// Seq2SeqAttn_17944373363321
// MI455X (gfx1250) — matrix operand order repaired and checked by reading, not compiled, not run on hardware
//
#include <hip/hip_runtime.h>

// ---------------------------------------------------------------------------
// Types for CDNA5 WMMA (wave32): v_wmma_f32_16x16x32_bf16
// ---------------------------------------------------------------------------
typedef __attribute__((ext_vector_type(16))) __bf16 v16bf;
typedef __attribute__((ext_vector_type(8)))  float  v8f;

union Frag { uint4 q[2]; v16bf v; };

#define NEGINF (-1.0e9f)

__device__ __forceinline__ unsigned short f2bf(float f) {
  unsigned u = __float_as_uint(f);
  u += 0x7FFFu + ((u >> 16) & 1u);   // round-to-nearest-even
  return (unsigned short)(u >> 16);
}
__device__ __forceinline__ float sigm(float x) { return 1.0f / (1.0f + __expf(-x)); }

// Async global->LDS copy, 16 bytes per lane (ASYNCcnt-tracked), CDNA5 TDM-lite path.
__device__ __forceinline__ void async_b128(unsigned lds_off, const void* gaddr) {
  asm volatile("global_load_async_to_lds_b128 %0, %1, off"
               :: "v"(lds_off), "v"(gaddr) : "memory");
}
__device__ __forceinline__ void wait_async_le4() {
  asm volatile("s_wait_asynccnt 0x4" ::: "memory");
}
__device__ __forceinline__ void wait_async_le0() {
  asm volatile("s_wait_asynccnt 0x0" ::: "memory");
}

// ---------------------------------------------------------------------------
// Elementwise / prep kernels
// ---------------------------------------------------------------------------
__global__ void k_cvt_bf16(const float* __restrict__ src, unsigned short* __restrict__ dst, int n) {
  int i = blockIdx.x * blockDim.x + threadIdx.x;
  if (i < n) dst[i] = f2bf(src[i]);
}

__global__ void k_fill0(unsigned int* __restrict__ p, int n32) {
  int i = blockIdx.x * blockDim.x + threadIdx.x;
  if (i < n32) p[i] = 0u;
}

// Wcat[n][0:1536]=Wih_d[n], Wcat[n][1536:2048]=Whh_d[n]  (2048x2048 bf16)
__global__ void k_concat_dec_w(const float* __restrict__ Wih, const float* __restrict__ Whh,
                               unsigned short* __restrict__ Wcat) {
  int i = blockIdx.x * blockDim.x + threadIdx.x;
  if (i >= 2048 * 2048) return;
  int n = i >> 11, k = i & 2047;
  float v = (k < 1536) ? Wih[n * 1536 + k] : Whh[n * 512 + (k - 1536)];
  Wcat[i] = f2bf(v);
}

// X[(s*16+b)*512+e] = bf16(enc_embed[src[b][s]][e])
__global__ void k_gather_enc(const float* __restrict__ emb, const int* __restrict__ src,
                             unsigned short* __restrict__ X) {
  int i = blockIdx.x * blockDim.x + threadIdx.x;
  if (i >= 2048 * 512) return;
  int r = i >> 9, e = i & 511;
  int s = r >> 4, b = r & 15;
  int tok = src[b * 128 + s];
  X[i] = f2bf(emb[(size_t)tok * 512 + e]);
}

// xin[b][0:512] = bf16(dec_embed[tgt[b][t]])
__global__ void k_gather_dec(const float* __restrict__ emb, const int* __restrict__ tgt,
                             int t, unsigned short* __restrict__ xin) {
  int i = blockIdx.x * blockDim.x + threadIdx.x;
  if (i >= 16 * 512) return;
  int b = i >> 9, e = i & 511;
  int tok = tgt[b * 128 + t];
  xin[b * 2048 + e] = f2bf(emb[(size_t)tok * 512 + e]);
}

// ---------------------------------------------------------------------------
// Big GEMM: C[M,N] = A[M,K](bf16) * W[N,K]^T(bf16) (+bias)
// Block tile 128x128, 8 waves, each wave 2x4 WMMA tiles, K step 32.
// Double-buffered LDS filled by async global->LDS (ASYNCcnt) loads.
// A must be row-padded so that ceil(M/128)*128 rows are readable.
// outPerm: row r=(t*16+b) -> C[(b*Tsteps+t)*ldc+col]  (final FC layout)
// ---------------------------------------------------------------------------
__global__ __launch_bounds__(256) void k_gemm_big(
    const unsigned short* __restrict__ A, int lda,
    const unsigned short* __restrict__ W,
    const float* __restrict__ bias,
    float* __restrict__ C, int ldc,
    int M, int K, int outPerm, int Tsteps) {
  __shared__ __align__(16) unsigned short As[2][128 * 32];
  __shared__ __align__(16) unsigned short Bs[2][128 * 32];
  const int tid  = threadIdx.x;
  const int lane = tid & 31;
  const int wv   = tid >> 5;
  const int h    = lane >> 4;
  const int l15  = lane & 15;
  const int bm   = blockIdx.y * 128;
  const int bn   = blockIdx.x * 128;
  const int wm   = (wv & 3) * 32;
  const int wn   = (wv >> 2) * 64;

  // Per-thread tile-fill coordinates: 512 uint4 per tile, 2 per thread.
  const int r0 = tid >> 2,           s0 = (tid & 3) * 8;
  const int r1 = (tid + 256) >> 2,   s1 = s0;
  const unsigned short* Ab = A + (size_t)bm * lda;
  const unsigned short* Wb = W + (size_t)bn * K;

  v8f acc[2][4];
#pragma unroll
  for (int a = 0; a < 2; a++)
#pragma unroll
    for (int b = 0; b < 4; b++)
#pragma unroll
      for (int q = 0; q < 8; q++) acc[a][b][q] = 0.0f;

  auto issue = [&](int buf, int k) {
    async_b128((unsigned)(size_t)&As[buf][r0 * 32 + s0], Ab + (size_t)r0 * lda + k + s0);
    async_b128((unsigned)(size_t)&As[buf][r1 * 32 + s1], Ab + (size_t)r1 * lda + k + s1);
    async_b128((unsigned)(size_t)&Bs[buf][r0 * 32 + s0], Wb + (size_t)r0 * K + k + s0);
    async_b128((unsigned)(size_t)&Bs[buf][r1 * 32 + s1], Wb + (size_t)r1 * K + k + s1);
  };
  auto compute = [&](int buf) {
    Frag af[2], bfr[4];
#pragma unroll
    for (int im = 0; im < 2; im++) {
      int r = wm + im * 16 + l15;
      af[im].q[0] = *reinterpret_cast<const uint4*>(&As[buf][r * 32 + h * 8]);
      af[im].q[1] = *reinterpret_cast<const uint4*>(&As[buf][r * 32 + 16 + h * 8]);
    }
#pragma unroll
    for (int in = 0; in < 4; in++) {
      int n = wn + in * 16 + l15;
      bfr[in].q[0] = *reinterpret_cast<const uint4*>(&Bs[buf][n * 32 + h * 8]);
      bfr[in].q[1] = *reinterpret_cast<const uint4*>(&Bs[buf][n * 32 + 16 + h * 8]);
    }
#pragma unroll
    for (int im = 0; im < 2; im++)
#pragma unroll
      for (int in = 0; in < 4; in++)
        acc[im][in] = __builtin_amdgcn_wmma_f32_16x16x32_bf16(
            false, af[im].v, false, bfr[in].v, (short)0, acc[im][in], false, false);
  };

  // Software pipeline: tile k+32 in flight while computing tile k.
  issue(0, 0);
  int cur = 0, k0 = 0;
  for (; k0 + 32 < K; k0 += 32, cur ^= 1) {
    issue(cur ^ 1, k0 + 32);
    wait_async_le4();        // this wave's tile-k loads (first 4 of 8) complete
    __syncthreads();
    compute(cur);
    __syncthreads();         // everyone done reading before buffer is refilled
  }
  wait_async_le0();
  __syncthreads();
  compute(cur);

#pragma unroll
  for (int im = 0; im < 2; im++) {
#pragma unroll
    for (int in = 0; in < 4; in++) {
      int col = bn + wn + in * 16 + l15;
      float bv = bias ? bias[col] : 0.0f;
#pragma unroll
      for (int r = 0; r < 8; r++) {
        int row = bm + wm + im * 16 + h * 8 + r;
        if (row < M) {
          float v = acc[im][in][r] + bv;
          size_t o;
          if (outPerm) {
            int b = row & 15, t = row >> 4;
            o = (size_t)(b * Tsteps + t) * ldc + col;
          } else {
            o = (size_t)row * ldc + col;
          }
          C[o] = v;
        }
      }
    }
  }
}

// ---------------------------------------------------------------------------
// Small-M GEMM: M=16 exactly (batch B=16). One wave per 16x16 C tile.
// C[16,N] fp32 (+bias, opt tanh), optional bf16 copy with its own stride.
// N must be a multiple of 128. A rows have stride lda (bf16 elems).
// ---------------------------------------------------------------------------
__global__ __launch_bounds__(256) void k_gemm_row16(
    const unsigned short* __restrict__ A, int lda,
    const unsigned short* __restrict__ W,
    const float* __restrict__ bias,
    float* __restrict__ C, int ldc,
    unsigned short* __restrict__ Cbf, int ldbf,
    int K, int doTanh) {
  const int tid  = threadIdx.x;
  const int lane = tid & 31;
  const int wv   = tid >> 5;
  const int h    = lane >> 4;
  const int l15  = lane & 15;
  const int n    = blockIdx.x * 128 + wv * 16 + l15;

  v8f acc;
#pragma unroll
  for (int q = 0; q < 8; q++) acc[q] = 0.0f;

  const unsigned short* Arow = A + (size_t)l15 * lda;
  const unsigned short* Wrow = W + (size_t)n * K;
  for (int k0 = 0; k0 < K; k0 += 32) {
    Frag af, bfr;
    af.q[0]  = *reinterpret_cast<const uint4*>(Arow + k0 + h * 8);
    af.q[1]  = *reinterpret_cast<const uint4*>(Arow + k0 + 16 + h * 8);
    bfr.q[0] = *reinterpret_cast<const uint4*>(Wrow + k0 + h * 8);
    bfr.q[1] = *reinterpret_cast<const uint4*>(Wrow + k0 + 16 + h * 8);
    acc = __builtin_amdgcn_wmma_f32_16x16x32_bf16(
        false, af.v, false, bfr.v, (short)0, acc, false, false);
  }

  float bv = bias ? bias[n] : 0.0f;
#pragma unroll
  for (int r = 0; r < 8; r++) {
    int m = h * 8 + r;
    float v = acc[r] + bv;
    if (doTanh) v = tanhf(v);
    if (C)   C[(size_t)m * ldc + n] = v;
    if (Cbf) Cbf[(size_t)m * ldbf + n] = f2bf(v);
  }
}

// ---------------------------------------------------------------------------
// LSTM cell: gates = preX (+preH) + bih + bhh ; i,f,g,o split.
// Writes c (in/out), and h to up to three destinations (fp32 + 2x bf16).
// ---------------------------------------------------------------------------
__global__ void k_lstm_cell(const float* __restrict__ preX, const float* __restrict__ preH,
                            const float* __restrict__ bih, const float* __restrict__ bhh,
                            float* __restrict__ c,
                            float* __restrict__ hF, int sF,
                            unsigned short* __restrict__ hB1, int s1,
                            unsigned short* __restrict__ hB2, int s2,
                            int Hd) {
  int i = blockIdx.x * blockDim.x + threadIdx.x;
  if (i >= 16 * Hd) return;
  int b = i / Hd, j = i % Hd;
  size_t base = (size_t)b * 4 * Hd;
  float gi = preX[base + j]          + bih[j]          + bhh[j];
  float gf = preX[base + Hd + j]     + bih[Hd + j]     + bhh[Hd + j];
  float gg = preX[base + 2 * Hd + j] + bih[2 * Hd + j] + bhh[2 * Hd + j];
  float go = preX[base + 3 * Hd + j] + bih[3 * Hd + j] + bhh[3 * Hd + j];
  if (preH) {
    gi += preH[base + j];
    gf += preH[base + Hd + j];
    gg += preH[base + 2 * Hd + j];
    go += preH[base + 3 * Hd + j];
  }
  float cv = sigm(gf) * c[i] + sigm(gi) * tanhf(gg);
  c[i] = cv;
  float hv = sigm(go) * tanhf(cv);
  if (hF)  hF[(size_t)b * sF + j]  = hv;
  if (hB1) hB1[(size_t)b * s1 + j] = f2bf(hv);
  if (hB2) hB2[(size_t)b * s2 + j] = f2bf(hv);
}

// ---------------------------------------------------------------------------
// Attention score: one wave per (b,s): sum_a v[a]*tanh(encp[b,s,a]+sproj[b,a])
// ---------------------------------------------------------------------------
__global__ __launch_bounds__(128) void k_attn_score(
    const float* __restrict__ encp, const float* __restrict__ sproj,
    const float* __restrict__ v, const int* __restrict__ lens,
    float* __restrict__ score) {
  int wid  = blockIdx.x * 4 + (threadIdx.x >> 5);
  int lane = threadIdx.x & 31;
  int b = wid >> 7, s = wid & 127;
  const float* ep = encp + (size_t)(b * 128 + s) * 512;
  const float* sp = sproj + (size_t)b * 512;
  float acc = 0.0f;
  for (int a = lane; a < 512; a += 32) acc += v[a] * tanhf(ep[a] + sp[a]);
#pragma unroll
  for (int off = 16; off; off >>= 1) acc += __shfl_xor(acc, off, 32);
  if (lane == 0) score[b * 128 + s] = (s < lens[b]) ? acc : NEGINF;
}

__global__ __launch_bounds__(128) void k_softmax(const float* __restrict__ score,
                                                 float* __restrict__ attn) {
  __shared__ float red[128];
  int b = blockIdx.x, t = threadIdx.x;
  float v = score[b * 128 + t];
  red[t] = v; __syncthreads();
  for (int off = 64; off; off >>= 1) { if (t < off) red[t] = fmaxf(red[t], red[t + off]); __syncthreads(); }
  float mx = red[0]; __syncthreads();
  float e = __expf(v - mx);
  red[t] = e; __syncthreads();
  for (int off = 64; off; off >>= 1) { if (t < off) red[t] += red[t + off]; __syncthreads(); }
  attn[b * 128 + t] = e / red[0];
}

// ctx[b,e] = sum_s attn[b,s]*enc_out[b,s,e]; write bf16 into xin & HC row t
__global__ void k_ctx(const float* __restrict__ attn, const float* __restrict__ enc,
                      unsigned short* __restrict__ xin, unsigned short* __restrict__ HC, int t) {
  int i = blockIdx.x * blockDim.x + threadIdx.x;
  if (i >= 16 * 1024) return;
  int b = i >> 10, e = i & 1023;
  const float* ar = attn + b * 128;
  const float* er = enc + (size_t)b * 128 * 1024 + e;
  float acc = 0.0f;
  for (int s = 0; s < 128; s++) acc += ar[s] * er[(size_t)s * 1024];
  unsigned short bv = f2bf(acc);
  xin[b * 2048 + 512 + e] = bv;
  HC[(size_t)(t * 16 + b) * 1536 + 512 + e] = bv;
}

// ---------------------------------------------------------------------------
// Host orchestration
// ---------------------------------------------------------------------------
extern "C" void kernel_launch(void* const* d_in, const int* in_sizes, int n_in,
                              void* d_out, int out_size, void* d_ws, size_t ws_size,
                              hipStream_t stream) {
  (void)in_sizes; (void)n_in; (void)out_size; (void)ws_size;
  const int S = 128, B = 16, Tm1 = 127;

  const int*   src      = (const int*)d_in[0];
  const int*   src_lens = (const int*)d_in[1];
  const int*   tgt      = (const int*)d_in[2];
  const float* enc_emb  = (const float*)d_in[3];
  const float* Wih_f = (const float*)d_in[4];
  const float* Whh_f = (const float*)d_in[5];
  const float* bih_f = (const float*)d_in[6];
  const float* bhh_f = (const float*)d_in[7];
  const float* Wih_b = (const float*)d_in[8];
  const float* Whh_b = (const float*)d_in[9];
  const float* bih_b = (const float*)d_in[10];
  const float* bhh_b = (const float*)d_in[11];
  const float* Wbh   = (const float*)d_in[12];
  const float* bbh   = (const float*)d_in[13];
  const float* Wbc   = (const float*)d_in[14];
  const float* bbc   = (const float*)d_in[15];
  const float* Wh_att = (const float*)d_in[16];
  const float* Ws_att = (const float*)d_in[17];
  const float* v_att  = (const float*)d_in[18];
  const float* dec_emb = (const float*)d_in[19];
  const float* Wih_d = (const float*)d_in[20];
  const float* Whh_d = (const float*)d_in[21];
  const float* bih_d = (const float*)d_in[22];
  const float* bhh_d = (const float*)d_in[23];
  const float* Wfc   = (const float*)d_in[24];
  const float* bfc   = (const float*)d_in[25];
  float* out = (float*)d_out;

  char* ws = (char*)d_ws;
  size_t off = 0;
  auto alloc = [&](size_t bytes) -> char* {
    char* p = ws + off;
    off += (bytes + 255) & ~(size_t)255;
    return p;
  };

  // bf16 weights
  unsigned short* Wihf = (unsigned short*)alloc((size_t)2048 * 512 * 2);
  unsigned short* Whhf = (unsigned short*)alloc((size_t)2048 * 512 * 2);
  unsigned short* Wihb = (unsigned short*)alloc((size_t)2048 * 512 * 2);
  unsigned short* Whhb = (unsigned short*)alloc((size_t)2048 * 512 * 2);
  unsigned short* Wbhb = (unsigned short*)alloc((size_t)512 * 1024 * 2);
  unsigned short* Wbcb = (unsigned short*)alloc((size_t)512 * 1024 * 2);
  unsigned short* Whab = (unsigned short*)alloc((size_t)512 * 1024 * 2);
  unsigned short* Wsab = (unsigned short*)alloc((size_t)512 * 512 * 2);
  unsigned short* Wcat = (unsigned short*)alloc((size_t)2048 * 2048 * 2);
  unsigned short* Wfcb = (unsigned short*)alloc((size_t)32000 * 1536 * 2);
  // activations (HC padded to 2048 rows so gemm_big can load unguarded)
  unsigned short* Xbf   = (unsigned short*)alloc((size_t)2048 * 512 * 2);
  unsigned short* encBf = (unsigned short*)alloc((size_t)2048 * 1024 * 2);
  unsigned short* HC    = (unsigned short*)alloc((size_t)2048 * 1536 * 2);
  unsigned short* xin   = (unsigned short*)alloc((size_t)16 * 2048 * 2);
  unsigned short* hcat  = (unsigned short*)alloc((size_t)16 * 1024 * 2);
  float* preF   = (float*)alloc((size_t)2048 * 2048 * 4);
  float* preB   = (float*)alloc((size_t)2048 * 2048 * 4);
  float* encOut = (float*)alloc((size_t)2048 * 1024 * 4);
  float* encPrj = (float*)alloc((size_t)2048 * 512 * 4);
  float* gates  = (float*)alloc((size_t)16 * 2048 * 4);
  float* c_f    = (float*)alloc((size_t)16 * 512 * 4);
  float* c_b    = (float*)alloc((size_t)16 * 512 * 4);
  float* c_d    = (float*)alloc((size_t)16 * 512 * 4);
  float* sproj  = (float*)alloc((size_t)16 * 512 * 4);
  float* score  = (float*)alloc((size_t)16 * 128 * 4);
  float* attn   = (float*)alloc((size_t)16 * 128 * 4);
  float* h0tmp  = (float*)alloc((size_t)16 * 512 * 4);

  auto cvt = [&](const float* s, unsigned short* d, int n) {
    k_cvt_bf16<<<(n + 255) / 256, 256, 0, stream>>>(s, d, n);
  };
  // weight conversion
  cvt(Wih_f, Wihf, 2048 * 512);  cvt(Whh_f, Whhf, 2048 * 512);
  cvt(Wih_b, Wihb, 2048 * 512);  cvt(Whh_b, Whhb, 2048 * 512);
  cvt(Wbh, Wbhb, 512 * 1024);    cvt(Wbc, Wbcb, 512 * 1024);
  cvt(Wh_att, Whab, 512 * 1024); cvt(Ws_att, Wsab, 512 * 512);
  cvt(Wfc, Wfcb, 32000 * 1536);
  k_concat_dec_w<<<(2048 * 2048 + 255) / 256, 256, 0, stream>>>(Wih_d, Whh_d, Wcat);

  // embeddings + zero state
  k_gather_enc<<<(2048 * 512 + 255) / 256, 256, 0, stream>>>(enc_emb, src, Xbf);
  k_fill0<<<(16 * 1024 / 2 + 255) / 256, 256, 0, stream>>>((unsigned int*)hcat, 16 * 1024 / 2);
  k_fill0<<<(16 * 512 + 255) / 256, 256, 0, stream>>>((unsigned int*)c_f, 16 * 512);
  k_fill0<<<(16 * 512 + 255) / 256, 256, 0, stream>>>((unsigned int*)c_b, 16 * 512);

  // Batched input projections: pre = X(2048x512) @ Wih^T -> (2048x2048)
  k_gemm_big<<<dim3(2048 / 128, 2048 / 128), 256, 0, stream>>>(
      Xbf, 512, Wihf, nullptr, preF, 2048, 2048, 512, 0, 0);
  k_gemm_big<<<dim3(2048 / 128, 2048 / 128), 256, 0, stream>>>(
      Xbf, 512, Wihb, nullptr, preB, 2048, 2048, 512, 0, 0);

  // Forward encoder recurrence
  for (int s = 0; s < S; s++) {
    k_gemm_row16<<<2048 / 128, 256, 0, stream>>>(
        hcat, 1024, Whhf, nullptr, gates, 2048, nullptr, 0, 512, 0);
    k_lstm_cell<<<(16 * 512 + 255) / 256, 256, 0, stream>>>(
        preF + (size_t)s * 16 * 2048, gates, bih_f, bhh_f, c_f,
        encOut + (size_t)s * 1024, 128 * 1024,
        encBf + (size_t)s * 1024, 128 * 1024,
        hcat, 1024, 512);
  }
  // Backward encoder recurrence (store at reversed time index)
  for (int t = 0; t < S; t++) {
    int ss = S - 1 - t;
    k_gemm_row16<<<2048 / 128, 256, 0, stream>>>(
        hcat + 512, 1024, Whhb, nullptr, gates, 2048, nullptr, 0, 512, 0);
    k_lstm_cell<<<(16 * 512 + 255) / 256, 256, 0, stream>>>(
        preB + (size_t)ss * 16 * 2048, gates, bih_b, bhh_b, c_b,
        encOut + (size_t)ss * 1024 + 512, 128 * 1024,
        encBf + (size_t)ss * 1024 + 512, 128 * 1024,
        hcat + 512, 1024, 512);
  }

  // h0 = tanh([hf,hb]@Wbh^T + bbh) -> bf16 into xin h-slot; c0 -> c_d (fp32)
  k_gemm_row16<<<512 / 128, 256, 0, stream>>>(
      hcat, 1024, Wbhb, bbh, h0tmp, 512, xin + 1536, 2048, 1024, 1);
  k_gemm_row16<<<512 / 128, 256, 0, stream>>>(
      hcat, 1024, Wbcb, bbc, c_d, 512, nullptr, 0, 1024, 1);

  // enc_proj = enc_out(2048x1024) @ Wh_att^T -> (2048x512)
  k_gemm_big<<<dim3(512 / 128, 2048 / 128), 256, 0, stream>>>(
      encBf, 1024, Whab, nullptr, encPrj, 512, 2048, 1024, 0, 0);

  // Decoder recurrence: store h,ctx rows into HC; defer the big FC
  for (int t = 0; t < Tm1; t++) {
    k_gather_dec<<<(16 * 512 + 255) / 256, 256, 0, stream>>>(dec_emb, tgt, t, xin);
    k_gemm_row16<<<512 / 128, 256, 0, stream>>>(
        xin + 1536, 2048, Wsab, nullptr, sproj, 512, nullptr, 0, 512, 0);
    k_attn_score<<<(B * S) / 4, 128, 0, stream>>>(encPrj, sproj, v_att, src_lens, score);
    k_softmax<<<B, 128, 0, stream>>>(score, attn);
    k_ctx<<<(16 * 1024 + 255) / 256, 256, 0, stream>>>(attn, encOut, xin, HC, t);
    k_gemm_row16<<<2048 / 128, 256, 0, stream>>>(
        xin, 2048, Wcat, nullptr, gates, 2048, nullptr, 0, 2048, 0);
    k_lstm_cell<<<(16 * 512 + 255) / 256, 256, 0, stream>>>(
        gates, nullptr, bih_d, bhh_d, c_d,
        nullptr, 0,
        xin + 1536, 2048,
        HC + (size_t)t * 16 * 1536, 1536, 512);
  }

  // One fused vocab projection: (2048-padded rows, M=2032) @ Wfc^T + bfc
  k_gemm_big<<<dim3(32000 / 128, 2048 / 128), 256, 0, stream>>>(
      HC, 1536, Wfcb, bfc, out, 32000, 2032, 1536, 1, Tm1);
}
